// Decoder_33672543600678
// MI455X (gfx1250) — matrix operand order repaired and checked by reading, not compiled, not run on hardware
//
#include <hip/hip_runtime.h>

// ---------------------------------------------------------------------------
// Types for WMMA fragments (CDNA5 / gfx1250, wave32)
// ---------------------------------------------------------------------------
typedef __attribute__((ext_vector_type(16))) __bf16        v16bf;
typedef __attribute__((ext_vector_type(8)))  float          v8f;
typedef __attribute__((ext_vector_type(16))) unsigned short v16u;
typedef __attribute__((ext_vector_type(8)))  unsigned short v8u;

__device__ __forceinline__ v16bf as_bf16(v16u u) {
  union { v16u u; v16bf b; } c; c.u = u; return c.b;
}

// f32 -> bf16 with round-to-nearest-even
__device__ __forceinline__ unsigned short f2bf(float f) {
  unsigned int u = __float_as_uint(f);
  unsigned int r = u + 0x7FFFu + ((u >> 16) & 1u);
  return (unsigned short)(r >> 16);
}

__device__ __forceinline__ float sigf(float x) {
  return 1.0f / (1.0f + __expf(-x));
}

// Problem sizes
#define SLEN 128
#define BATCH 256
#define HID 512
#define H3 1536
#define EMB 256
#define VOCAB 32000
#define KCAT 2304     // E + 3H + H (fused [x | h0] K dim for the gates GEMM)
#define LDA_ATT 1544  // 1536 + 8 bf16 pad -> conflict-free LDS banks

// ---------------------------------------------------------------------------
// Conversion kernels (weights stay L2-resident afterwards)
// ---------------------------------------------------------------------------
__global__ void f32_to_bf16_kernel(const float* __restrict__ src,
                                   unsigned short* __restrict__ dst, int n) {
  int i = blockIdx.x * blockDim.x + threadIdx.x;
  if (i < n) dst[i] = f2bf(src[i]);
}

// Fuse W_ih[2048,1792] and W_hh[2048,512] into Wc[2048,2304] bf16
__global__ void build_wc_kernel(const float* __restrict__ wih,
                                const float* __restrict__ whh,
                                unsigned short* __restrict__ dst) {
  int i = blockIdx.x * blockDim.x + threadIdx.x;
  if (i >= 2048 * KCAT) return;
  int row = i / KCAT, col = i - row * KCAT;
  float v = (col < 1792) ? wih[(size_t)row * 1792 + col]
                         : whh[(size_t)row * 512 + (col - 1792)];
  dst[i] = f2bf(v);
}

// ---------------------------------------------------------------------------
// Attention scorer: scores[s*B+b] = W2 . tanh(W1 . ctx(s,b) + b1) + b2
// Block = 8 waves over a 32-row M tile: 2 m-slices x 4 n-groups (24 N-tiles
// each). A tile staged once in dynamic LDS (bf16, padded rows). 98.8 KB LDS
// -> 3 workgroups / WGP -> 6 waves per SIMD of WMMA+ds_load pipelining.
// ---------------------------------------------------------------------------
__global__ __launch_bounds__(256) void attn_score_kernel(
    const float* __restrict__ enc,            // [S,B,1024]
    const float* __restrict__ hid,            // [B,512]
    const unsigned short* __restrict__ w1b,   // [1536,1536] bf16
    const float* __restrict__ b1,             // [1536]
    const float* __restrict__ w2,             // [1536]
    const float* __restrict__ b2,             // [1]
    float* __restrict__ scores)               // [S*B]
{
  extern __shared__ unsigned short At[];      // 32 * LDA_ATT ushorts (98,816 B)
  __shared__ float partial[4][32];            // n-group partial scores per row

  const int tid = threadIdx.x;
  const int s  = blockIdx.x >> 3;             // 8 m-tiles of 32 per s (B/32)
  const int b0 = (blockIdx.x & 7) << 5;

  // Stage A tile: rows = b0..b0+31 of [enc(s,b,:) | hid(b,:)], f32 -> bf16
  for (int idx = tid; idx < 32 * 384; idx += 256) {
    int r = idx / 384, c4 = idx - r * 384;    // c4 in float4 units
    float4 v;
    if (c4 < 256)
      v = *(const float4*)(enc + ((size_t)(s * BATCH + b0 + r)) * 1024 + c4 * 4);
    else
      v = *(const float4*)(hid + (size_t)(b0 + r) * HID + (c4 - 256) * 4);
    ushort4 o;
    o.x = f2bf(v.x); o.y = f2bf(v.y); o.z = f2bf(v.z); o.w = f2bf(v.w);
    *(ushort4*)(&At[r * LDA_ATT + c4 * 4]) = o;
  }
  __syncthreads();

  const int lane = tid & 31, wave = tid >> 5;
  const int slice = wave & 1;                 // which 16-row slice
  const int ngrp  = wave >> 1;                // which 24 N-tiles
  const int lq = lane & 15, hf = lane >> 4;

  float acc[8];
#pragma unroll
  for (int r = 0; r < 8; ++r) acc[r] = 0.0f;

  const unsigned short* arow = &At[(slice * 16 + lq) * LDA_ATT];

  for (int n = ngrp * 24; n < ngrp * 24 + 24; ++n) {
    const int ncol = n * 16 + lq;
    const unsigned short* brow = w1b + (size_t)ncol * H3;
    v8f c = {0.f, 0.f, 0.f, 0.f, 0.f, 0.f, 0.f, 0.f};
    for (int k = 0; k < H3; k += 32) {
      v8u alo = *(const v8u*)(arow + k + hf * 8);
      v8u ahi = *(const v8u*)(arow + k + 16 + hf * 8);
      v16u a  = __builtin_shufflevector(alo, ahi, 0, 1, 2, 3, 4, 5, 6, 7,
                                        8, 9, 10, 11, 12, 13, 14, 15);
      v8u blo = *(const v8u*)(brow + k + hf * 8);
      v8u bhi = *(const v8u*)(brow + k + 16 + hf * 8);
      v16u bb = __builtin_shufflevector(blo, bhi, 0, 1, 2, 3, 4, 5, 6, 7,
                                        8, 9, 10, 11, 12, 13, 14, 15);
      c = __builtin_amdgcn_wmma_f32_16x16x32_bf16(
              false, as_bf16(a), false, as_bf16(bb), (short)0, c, false, false);
    }
    float bias = b1[ncol], wv = w2[ncol];
#pragma unroll
    for (int r = 0; r < 8; ++r) acc[r] += tanhf(c[r] + bias) * wv;
  }

  // Reduce over the 16 N-lanes of each half-wave
#pragma unroll
  for (int r = 0; r < 8; ++r) {
    acc[r] += __shfl_xor(acc[r], 1);
    acc[r] += __shfl_xor(acc[r], 2);
    acc[r] += __shfl_xor(acc[r], 4);
    acc[r] += __shfl_xor(acc[r], 8);
  }
  if (lq == 0) {
#pragma unroll
    for (int r = 0; r < 8; ++r)                // C layout: row = r + 8*(lane/16)
      partial[ngrp][slice * 16 + hf * 8 + r] = acc[r];
  }
  __syncthreads();

  // Combine the 4 n-group partials
  if (tid < 32) {
    float v = partial[0][tid] + partial[1][tid] + partial[2][tid] +
              partial[3][tid] + b2[0];
    scores[s * BATCH + b0 + tid] = v;
  }
}

// ---------------------------------------------------------------------------
// Softmax over S + attention-weighted context + build fused RNN input
// xcat[b] = bf16([ emb[id[b]] (256) | context (1536) | h0[b] (512) ])
// context[1024:1536] == hidden[b] exactly (softmax sums to 1).
// ---------------------------------------------------------------------------
__global__ __launch_bounds__(256) void ctx_kernel(
    const int* __restrict__ ids, const float* __restrict__ enc,
    const float* __restrict__ hid, const float* __restrict__ emb,
    const float* __restrict__ scores, unsigned short* __restrict__ xcat)
{
  __shared__ float att[SLEN];
  __shared__ float red[SLEN];
  const int b = blockIdx.x, t = threadIdx.x;

  if (t < SLEN) { float v = scores[t * BATCH + b]; att[t] = v; red[t] = v; }
  __syncthreads();
  for (int off = 64; off > 0; off >>= 1) {
    if (t < off) red[t] = fmaxf(red[t], red[t + off]);
    __syncthreads();
  }
  float mx = red[0];
  __syncthreads();
  if (t < SLEN) { float e = __expf(att[t] - mx); att[t] = e; red[t] = e; }
  __syncthreads();
  for (int off = 64; off > 0; off >>= 1) {
    if (t < off) red[t] += red[t + off];
    __syncthreads();
  }
  float inv = 1.0f / red[0];
  __syncthreads();
  if (t < SLEN) att[t] *= inv;
  __syncthreads();

  unsigned short* xr = xcat + (size_t)b * KCAT;
  // embedded (E == blockDim)
  xr[t] = f2bf(emb[(size_t)ids[b] * EMB + t]);
  // context over encoder part (k < 1024)
  for (int i = 0; i < 4; ++i) {
    int k = t + 256 * i;
    const float* ep = enc + (size_t)b * 1024 + k;
    float a = 0.0f;
#pragma unroll 4
    for (int ss = 0; ss < SLEN; ++ss) a += att[ss] * ep[(size_t)ss * (BATCH * 1024)];
    xr[EMB + k] = f2bf(a);
  }
  // context hidden-slice == hidden[b];  h0 tail for the W_hh part
  for (int i = 0; i < 2; ++i) {
    int j = t + 256 * i;
    unsigned short hv = f2bf(hid[(size_t)b * HID + j]);
    xr[EMB + 1024 + j] = hv;   // 1280..1791
    xr[1792 + j]       = hv;   // 1792..2303
  }
}

// ---------------------------------------------------------------------------
// Generic bf16 WMMA GEMM: out[M,N] = A[M,K] * W[N,K]^T + bias1 (+bias2)
// block = 4 waves, each wave owns one 16x16 output tile.
// grid = (N/64, M/16)
// ---------------------------------------------------------------------------
__global__ __launch_bounds__(128) void gemm_bf16_kernel(
    const unsigned short* __restrict__ A, const unsigned short* __restrict__ W,
    const float* __restrict__ bias1, const float* __restrict__ bias2,
    float* __restrict__ out, int K, int ldA, int ldW, int ldC)
{
  const int lane = threadIdx.x & 31;
  const int wave = threadIdx.x >> 5;
  const int lq = lane & 15, hf = lane >> 4;
  const int n0 = (blockIdx.x * 4 + wave) * 16;
  const int m0 = blockIdx.y * 16;

  const unsigned short* Ar = A + (size_t)(m0 + lq) * ldA;
  const unsigned short* Wr = W + (size_t)(n0 + lq) * ldW;

  v8f c = {0.f, 0.f, 0.f, 0.f, 0.f, 0.f, 0.f, 0.f};
#pragma unroll 4
  for (int k = 0; k < K; k += 32) {
    v8u alo = *(const v8u*)(Ar + k + hf * 8);
    v8u ahi = *(const v8u*)(Ar + k + 16 + hf * 8);
    v16u a  = __builtin_shufflevector(alo, ahi, 0, 1, 2, 3, 4, 5, 6, 7,
                                      8, 9, 10, 11, 12, 13, 14, 15);
    v8u blo = *(const v8u*)(Wr + k + hf * 8);
    v8u bhi = *(const v8u*)(Wr + k + 16 + hf * 8);
    v16u bb = __builtin_shufflevector(blo, bhi, 0, 1, 2, 3, 4, 5, 6, 7,
                                      8, 9, 10, 11, 12, 13, 14, 15);
    c = __builtin_amdgcn_wmma_f32_16x16x32_bf16(
            false, as_bf16(a), false, as_bf16(bb), (short)0, c, false, false);
  }

  const int n = n0 + lq;
  float bias = bias1 ? bias1[n] : 0.0f;
  if (bias2) bias += bias2[n];
#pragma unroll
  for (int r = 0; r < 8; ++r)
    out[(size_t)(m0 + hf * 8 + r) * ldC + n] = c[r] + bias;
}

// ---------------------------------------------------------------------------
// LSTM cell elementwise (PyTorch i,f,g,o gate order)
// ---------------------------------------------------------------------------
__global__ __launch_bounds__(256) void lstm_cell_kernel(
    const float* __restrict__ gates, const float* __restrict__ cell0,
    float* __restrict__ hout, float* __restrict__ cout,
    unsigned short* __restrict__ hb)
{
  int idx = blockIdx.x * blockDim.x + threadIdx.x;  // < B*H
  int b = idx >> 9, j = idx & 511;
  const float* g = gates + (size_t)b * 2048;
  float ig = g[j], fg = g[512 + j], gg = g[1024 + j], og = g[1536 + j];
  float c0 = cell0[idx];
  float cn = sigf(fg) * c0 + sigf(ig) * tanhf(gg);
  float hn = sigf(og) * tanhf(cn);
  hout[idx] = hn;
  cout[idx] = cn;
  hb[idx] = f2bf(hn);
}

// ---------------------------------------------------------------------------
// Launch
// ---------------------------------------------------------------------------
static inline size_t align256(size_t x) { return (x + 255) & ~(size_t)255; }

extern "C" void kernel_launch(void* const* d_in, const int* in_sizes, int n_in,
                              void* d_out, int out_size, void* d_ws, size_t ws_size,
                              hipStream_t stream) {
  (void)in_sizes; (void)n_in; (void)out_size; (void)ws_size;
  const int*   ids   = (const int*)  d_in[0];
  const float* enc   = (const float*)d_in[1];
  const float* hid   = (const float*)d_in[2];
  const float* cell0 = (const float*)d_in[3];
  const float* emb   = (const float*)d_in[4];
  const float* wih   = (const float*)d_in[5];
  const float* whh   = (const float*)d_in[6];
  const float* bih   = (const float*)d_in[7];
  const float* bhh   = (const float*)d_in[8];
  const float* w1    = (const float*)d_in[9];
  const float* b1    = (const float*)d_in[10];
  const float* w2    = (const float*)d_in[11];
  const float* b2    = (const float*)d_in[12];
  const float* outw  = (const float*)d_in[13];
  const float* outb  = (const float*)d_in[14];
  float* out = (float*)d_out;

  // Workspace carve-up
  char* ws = (char*)d_ws;
  size_t o = 0;
  unsigned short* w1b = (unsigned short*)(ws + o); o = align256(o + (size_t)H3 * H3 * 2);
  unsigned short* wcb = (unsigned short*)(ws + o); o = align256(o + (size_t)2048 * KCAT * 2);
  unsigned short* owb = (unsigned short*)(ws + o); o = align256(o + (size_t)VOCAB * HID * 2);
  float* scores       = (float*)(ws + o);          o = align256(o + (size_t)SLEN * BATCH * 4);
  unsigned short* xcat= (unsigned short*)(ws + o); o = align256(o + (size_t)BATCH * KCAT * 2);
  float* gates        = (float*)(ws + o);          o = align256(o + (size_t)BATCH * 2048 * 4);
  unsigned short* hb  = (unsigned short*)(ws + o); o = align256(o + (size_t)BATCH * HID * 2);

  // 1) Weight conversions (bf16, L2-resident thereafter)
  {
    int n = H3 * H3;
    f32_to_bf16_kernel<<<(n + 255) / 256, 256, 0, stream>>>(w1, w1b, n);
  }
  {
    int n = 2048 * KCAT;
    build_wc_kernel<<<(n + 255) / 256, 256, 0, stream>>>(wih, whh, wcb);
  }
  {
    int n = VOCAB * HID;
    f32_to_bf16_kernel<<<(n + 255) / 256, 256, 0, stream>>>(outw, owb, n);
  }

  // 2) Attention scorer GEMM (dominant: 155 GFLOP, WMMA bf16)
  //    1024 blocks x 8 waves, 32-row M tiles, 98.8 KB dynamic LDS each.
  {
    size_t lds_bytes = (size_t)32 * LDA_ATT * sizeof(unsigned short);
    attn_score_kernel<<<(SLEN * BATCH) / 32, 256, lds_bytes, stream>>>(
        enc, hid, w1b, b1, w2, b2, scores);
  }

  // 3) Softmax + context + fused RNN input
  ctx_kernel<<<BATCH, 256, 0, stream>>>(ids, enc, hid, emb, scores, xcat);

  // 4) Gates GEMM: [256,2304] x [2048,2304]^T (+ b_ih + b_hh)
  gemm_bf16_kernel<<<dim3(2048 / 64, BATCH / 16), 128, 0, stream>>>(
      xcat, wcb, bih, bhh, gates, KCAT, KCAT, KCAT, 2048);

  // 5) LSTM cell -> h_new (f32 to d_out, bf16 to ws), c_new
  float* h_out = out + (size_t)BATCH * VOCAB;
  float* c_out = h_out + (size_t)BATCH * HID;
  lstm_cell_kernel<<<(BATCH * HID) / 256, 256, 0, stream>>>(
      gates, cell0, h_out, c_out, hb);

  // 6) Logits GEMM: [256,512] x [32000,512]^T + out_b -> d_out
  gemm_bf16_kernel<<<dim3(VOCAB / 64, BATCH / 16), 128, 0, stream>>>(
      hb, owb, outb, nullptr, out, HID, HID, HID, VOCAB);
}
